// GraphAttentionLayer_13477607375468
// MI455X (gfx1250) — compile-verified, matrix operand order checked by reading, not run on hardware
//
#include <hip/hip_runtime.h>
#include <hip/hip_bf16.h>

typedef __attribute__((ext_vector_type(16))) _Float16 v16h;
typedef __attribute__((ext_vector_type(8)))  _Float16 v8h;
typedef __attribute__((ext_vector_type(4)))  _Float16 v4h;
typedef __attribute__((ext_vector_type(8)))  float    v8f;

#define IN_F   512
#define OUT_F  256
#define NROW   8192
#define ALPHA  0.2f
#define NEG_BIG (-9.0e15f)

// ---------------------------------------------------------------------------
// Kernel 0a: convert x (f32) -> xh (f16), 4 elements per thread.
// ---------------------------------------------------------------------------
__global__ void gat_convert_x(const float* __restrict__ x, _Float16* __restrict__ xh) {
  const int gid = blockIdx.x * 256 + threadIdx.x;       // NROW*IN_F/4 threads
  const float4 v = ((const float4*)x)[gid];
  v4h o;
  o[0] = (_Float16)v.x; o[1] = (_Float16)v.y;
  o[2] = (_Float16)v.z; o[3] = (_Float16)v.w;
  ((v4h*)xh)[gid] = o;
}

// ---------------------------------------------------------------------------
// Kernel 0b: WTh[n][k] = (f16) W[k][n]  (256 x 512 transposed copy).
// ---------------------------------------------------------------------------
__global__ void gat_convert_wt(const float* __restrict__ W, _Float16* __restrict__ WTh) {
  const int gid = blockIdx.x * 256 + threadIdx.x;       // OUT_F*IN_F threads
  const int n = gid >> 9;          // /IN_F
  const int k = gid & (IN_F - 1);
  WTh[gid] = (_Float16)W[(size_t)k * OUT_F + n];
}

// ---------------------------------------------------------------------------
// Kernel 1: h = x @ W. Each wave computes a 16x32 tile (2 WMMAs / k-step).
// All operands pre-converted f16 -> inner loop is pure b128 loads + wmma.
// Also writes hT (f16, 256 x 8192) for the aggregation kernel's B operand.
// ---------------------------------------------------------------------------
__global__ void gat_gemm1(const _Float16* __restrict__ xh, const _Float16* __restrict__ WTh,
                          float* __restrict__ h, _Float16* __restrict__ hT) {
  const int tid  = threadIdx.x;
  const int lane = tid & 31;
  const int wave = tid >> 5;
  const int id   = blockIdx.x * 8 + wave;   // 4096 wave-tiles (512 M x 8 Npairs)
  const int i0 = (id >> 3) * 16;
  const int n0 = (id & 7) * 32;
  const int r  = lane & 15;
  const int hi = lane >> 4;
  const int c  = lane & 15;

  v8f acc0 = {}, acc1 = {};
#pragma unroll 4
  for (int k0 = 0; k0 < IN_F; k0 += 32) {
    const _Float16* ap = xh + (size_t)(i0 + r) * IN_F + k0 + hi * 8;
    v8h alo = *(const v8h*)ap;
    v8h ahi = *(const v8h*)(ap + 16);
    v16h A;
#pragma unroll
    for (int t = 0; t < 8; ++t) { A[t] = alo[t]; A[t + 8] = ahi[t]; }

    const _Float16* bp0 = WTh + (size_t)(n0 + c)      * IN_F + k0 + hi * 8;
    const _Float16* bp1 = WTh + (size_t)(n0 + 16 + c) * IN_F + k0 + hi * 8;
    v8h b0lo = *(const v8h*)bp0, b0hi = *(const v8h*)(bp0 + 16);
    v8h b1lo = *(const v8h*)bp1, b1hi = *(const v8h*)(bp1 + 16);
    v16h B0, B1;
#pragma unroll
    for (int t = 0; t < 8; ++t) {
      B0[t] = b0lo[t]; B0[t + 8] = b0hi[t];
      B1[t] = b1lo[t]; B1[t + 8] = b1hi[t];
    }
    acc0 = __builtin_amdgcn_wmma_f32_16x16x32_f16(false, A, false, B0, (short)0, acc0, false, false);
    acc1 = __builtin_amdgcn_wmma_f32_16x16x32_f16(false, A, false, B1, (short)0, acc1, false, false);
  }
#pragma unroll
  for (int v = 0; v < 8; ++v) {
    const int m = v + 8 * hi;
    const float v0 = acc0[v], v1 = acc1[v];
    h [(size_t)(i0 + m) * OUT_F + n0 + c]       = v0;
    h [(size_t)(i0 + m) * OUT_F + n0 + 16 + c]  = v1;
    hT[(size_t)(n0 + c)      * NROW + i0 + m]   = (_Float16)v0;
    hT[(size_t)(n0 + 16 + c) * NROW + i0 + m]   = (_Float16)v1;
  }
}

// ---------------------------------------------------------------------------
// Kernel 2: s1 = h @ a[:256], s2 = h @ a[256:]. One thread per row.
// ---------------------------------------------------------------------------
__global__ void gat_s12(const float* __restrict__ h, const float* __restrict__ a,
                        float* __restrict__ s1, float* __restrict__ s2) {
  const int i = blockIdx.x * blockDim.x + threadIdx.x;
  if (i >= NROW) return;
  const float* hp = h + (size_t)i * OUT_F;
  float acc1 = 0.f, acc2 = 0.f;
#pragma unroll 8
  for (int n = 0; n < OUT_F; ++n) {
    const float hv = hp[n];
    acc1 += hv * a[n];
    acc2 += hv * a[OUT_F + n];
  }
  s1[i] = acc1;
  s2[i] = acc2;
}

// ---------------------------------------------------------------------------
// Kernel 3: per-row softmax stats (max + sum of exp). adj read exactly once;
// per-thread e values stay in VGPRs. All-masked rows fall out naturally
// (m = NEG_BIG -> all weights 1 -> uniform softmax, matching the reference).
// ---------------------------------------------------------------------------
__global__ void gat_rowstats(const int* __restrict__ adj, const float* __restrict__ s1,
                             const float* __restrict__ s2, float* __restrict__ mrow,
                             float* __restrict__ zrow) {
  const int row = blockIdx.x;
  const int tid = threadIdx.x;
  const float s1v = s1[row];
  const int* ap = adj + (size_t)row * NROW;

  float ev[32];
  float lmax = NEG_BIG;
#pragma unroll
  for (int t = 0; t < 32; ++t) {
    const int j = t * 256 + tid;
    const float u  = s1v + s2[j];
    const float lr = fmaxf(u, ALPHA * u);
    const float e  = (ap[j] > 0) ? lr : NEG_BIG;
    ev[t] = e;
    lmax = fmaxf(lmax, e);
  }
  __shared__ float redm[8];
  __shared__ float redz[8];
#pragma unroll
  for (int o = 16; o > 0; o >>= 1) lmax = fmaxf(lmax, __shfl_down(lmax, o));
  if ((tid & 31) == 0) redm[tid >> 5] = lmax;
  __syncthreads();
  if (tid < 32) {
    float v = (tid < 8) ? redm[tid] : NEG_BIG;
#pragma unroll
    for (int o = 4; o > 0; o >>= 1) v = fmaxf(v, __shfl_down(v, o));
    if (tid == 0) redm[0] = v;
  }
  __syncthreads();
  const float m = redm[0];

  float lsum = 0.f;
#pragma unroll
  for (int t = 0; t < 32; ++t) lsum += __expf(ev[t] - m);
#pragma unroll
  for (int o = 16; o > 0; o >>= 1) lsum += __shfl_down(lsum, o);
  if ((tid & 31) == 0) redz[tid >> 5] = lsum;
  __syncthreads();
  if (tid == 0) {
    float z = 0.f;
#pragma unroll
    for (int w = 0; w < 8; ++w) z += redz[w];
    mrow[row] = m;
    zrow[row] = z;
  }
}

// ---------------------------------------------------------------------------
// Kernel 4: out = softmax(e) @ h + bias. Block = 32 rows x 256 cols; 8 waves,
// each owning 2 M-tiles x 2 N-tiles. K-step = 64: one ping-pong LDS stage +
// one barrier per 8 WMMAs. The 32x64 f16 weight tile (mask + lrelu + exp,
// f32 math) is staged by all 256 threads (8 elements each, vectorized loads,
// one v8h LDS store); the adj HBM stream is prefetched 512 columns ahead.
// ---------------------------------------------------------------------------
__global__ void gat_aggregate(const int* __restrict__ adj, const float* __restrict__ s1,
                              const float* __restrict__ s2, const float* __restrict__ mrow,
                              const float* __restrict__ zrow, const _Float16* __restrict__ hT,
                              const float* __restrict__ bias, float* __restrict__ out) {
  const int tid  = threadIdx.x;
  const int lane = tid & 31;
  const int wave = tid >> 5;
  const int i0 = blockIdx.x * 32;
  const int n0 = wave * 32;

  __shared__ __align__(16) _Float16 lds_w[2][32 * 64];   // 8 KB ping-pong
  __shared__ float s1l[32], ml[32], izl[32];

  if (tid < 32) {
    s1l[tid] = s1[i0 + tid];
    ml[tid]  = mrow[i0 + tid];
    izl[tid] = 1.0f / zrow[i0 + tid];
  }
  __syncthreads();

  const int r  = lane & 15;
  const int hi = lane >> 4;
  const int c  = lane & 15;
  const int srow = tid >> 3;        // staging row 0..31
  const int sj   = (tid & 7) * 8;   // staging col, 8 elems per thread

  const float s1s = s1l[srow];
  const float ms  = ml[srow];
  const int* arow = adj + (size_t)(i0 + srow) * NROW;

  auto stage = [&](int k0, int buf) {
    const int j = k0 + sj;
    const int4   a4a = *(const int4*)  (arow + j);
    const int4   a4b = *(const int4*)  (arow + j + 4);
    const float4 t4a = *(const float4*)(s2   + j);
    const float4 t4b = *(const float4*)(s2   + j + 4);
    float u[8] = { s1s + t4a.x, s1s + t4a.y, s1s + t4a.z, s1s + t4a.w,
                   s1s + t4b.x, s1s + t4b.y, s1s + t4b.z, s1s + t4b.w };
    const int av[8] = { a4a.x, a4a.y, a4a.z, a4a.w, a4b.x, a4b.y, a4b.z, a4b.w };
    v8h w;
#pragma unroll
    for (int t = 0; t < 8; ++t) {
      const float lr = fmaxf(u[t], ALPHA * u[t]);
      const float e  = (av[t] > 0) ? lr : NEG_BIG;
      w[t] = (_Float16)__expf(e - ms);
    }
    *(v8h*)&lds_w[buf][srow * 64 + sj] = w;
  };

  v8f acc00 = {}, acc01 = {}, acc10 = {}, acc11 = {};

  stage(0, 0);
  __syncthreads();

  const int NSTEP = NROW / 64;   // 128
  for (int s = 0; s < NSTEP; ++s) {
    const int k0  = s * 64;
    const int buf = s & 1;

    // Stage the next 32x64 weight tile into the other buffer (overlaps WMMA).
    if (s + 1 < NSTEP) stage(k0 + 64, buf ^ 1);
    // Prefetch the adjacency HBM stream 512 columns (8 steps) ahead.
    if (k0 + 512 + 64 <= NROW) __builtin_prefetch(arow + k0 + sj + 512, 0, 1);

    const _Float16* wt = lds_w[buf];
#pragma unroll
    for (int kk = 0; kk < 64; kk += 32) {
      // A tiles (rows 0-15 and 16-31) from LDS; row stride 64 halves.
      v8h a0lo = *(const v8h*)&wt[r * 64 + kk + hi * 8];
      v8h a0hi = *(const v8h*)&wt[r * 64 + kk + 16 + hi * 8];
      v8h a1lo = *(const v8h*)&wt[(16 + r) * 64 + kk + hi * 8];
      v8h a1hi = *(const v8h*)&wt[(16 + r) * 64 + kk + 16 + hi * 8];
      v16h A0, A1;
#pragma unroll
      for (int t = 0; t < 8; ++t) {
        A0[t] = a0lo[t]; A0[t + 8] = a0hi[t];
        A1[t] = a1lo[t]; A1[t + 8] = a1hi[t];
      }

      // B tiles from transposed f16 h (contiguous 16B loads).
      const _Float16* bp0 = hT + (size_t)(n0 + c)      * NROW + k0 + kk + hi * 8;
      const _Float16* bp1 = hT + (size_t)(n0 + 16 + c) * NROW + k0 + kk + hi * 8;
      v8h b0lo = *(const v8h*)bp0, b0hi = *(const v8h*)(bp0 + 16);
      v8h b1lo = *(const v8h*)bp1, b1hi = *(const v8h*)(bp1 + 16);
      v16h B0, B1;
#pragma unroll
      for (int t = 0; t < 8; ++t) {
        B0[t] = b0lo[t]; B0[t + 8] = b0hi[t];
        B1[t] = b1lo[t]; B1[t + 8] = b1hi[t];
      }

      acc00 = __builtin_amdgcn_wmma_f32_16x16x32_f16(false, A0, false, B0, (short)0, acc00, false, false);
      acc01 = __builtin_amdgcn_wmma_f32_16x16x32_f16(false, A0, false, B1, (short)0, acc01, false, false);
      acc10 = __builtin_amdgcn_wmma_f32_16x16x32_f16(false, A1, false, B0, (short)0, acc10, false, false);
      acc11 = __builtin_amdgcn_wmma_f32_16x16x32_f16(false, A1, false, B1, (short)0, acc11, false, false);
    }

    __syncthreads();
  }

#pragma unroll
  for (int v = 0; v < 8; ++v) {
    const int m = v + 8 * hi;
    const float iz0 = izl[m], iz1 = izl[16 + m];
    const float b0 = bias[n0 + c], b1 = bias[n0 + 16 + c];
    out[(size_t)(i0 + m)      * OUT_F + n0 + c]      = acc00[v] * iz0 + b0;
    out[(size_t)(i0 + m)      * OUT_F + n0 + 16 + c] = acc01[v] * iz0 + b1;
    out[(size_t)(i0 + 16 + m) * OUT_F + n0 + c]      = acc10[v] * iz1 + b0;
    out[(size_t)(i0 + 16 + m) * OUT_F + n0 + 16 + c] = acc11[v] * iz1 + b1;
  }
}

// ---------------------------------------------------------------------------
extern "C" void kernel_launch(void* const* d_in, const int* in_sizes, int n_in,
                              void* d_out, int out_size, void* d_ws, size_t ws_size,
                              hipStream_t stream) {
  (void)in_sizes; (void)n_in; (void)out_size; (void)ws_size;
  const float* x    = (const float*)d_in[0];
  const int*   adj  = (const int*)d_in[1];
  const float* W    = (const float*)d_in[2];
  const float* a    = (const float*)d_in[3];
  const float* bias = (const float*)d_in[4];
  float* out = (float*)d_out;

  char* ws = (char*)d_ws;
  float*    h   = (float*)ws;    ws += (size_t)NROW * OUT_F * sizeof(float);
  _Float16* hT  = (_Float16*)ws; ws += (size_t)NROW * OUT_F * sizeof(_Float16);
  _Float16* xh  = (_Float16*)ws; ws += (size_t)NROW * IN_F  * sizeof(_Float16);
  _Float16* WTh = (_Float16*)ws; ws += (size_t)OUT_F * IN_F * sizeof(_Float16);
  float* s1   = (float*)ws; ws += NROW * sizeof(float);
  float* s2   = (float*)ws; ws += NROW * sizeof(float);
  float* mrow = (float*)ws; ws += NROW * sizeof(float);
  float* zrow = (float*)ws; ws += NROW * sizeof(float);

  gat_convert_x <<<(NROW * IN_F / 4) / 256, 256, 0, stream>>>(x, xh);
  gat_convert_wt<<<(OUT_F * IN_F) / 256,    256, 0, stream>>>(W, WTh);
  gat_gemm1     <<<512,        256, 0, stream>>>(xh, WTh, h, hT);
  gat_s12       <<<NROW / 256, 256, 0, stream>>>(h, a, s1, s2);
  gat_rowstats  <<<NROW,       256, 0, stream>>>(adj, s1, s2, mrow, zrow);
  gat_aggregate <<<NROW / 32,  256, 0, stream>>>(adj, s1, s2, mrow, zrow, hT, bias, out);
}
